// MultiheadAttention_51281909514607
// MI455X (gfx1250) — matrix operand order repaired and checked by reading, not compiled, not run on hardware
//
#include <hip/hip_runtime.h>
#include <hip/hip_bf16.h>

// ---------------------------------------------------------------------------
// MHA fwd (B=2, T=1024, E=1024, H=16, HD=64) with XPOS rotary, causal mask,
// rel_pos bias, softmax. All GEMM-shaped work via v_wmma_f32_16x16x32_bf16.
// GEMM B-slabs staged to LDS with gfx1250 async-to-LDS (double buffered).
// ---------------------------------------------------------------------------

#define B_  2
#define T_  1024
#define E_  1024
#define H_  16
#define HD_ 64

typedef __attribute__((ext_vector_type(16))) __bf16 v16bf;
typedef __attribute__((ext_vector_type(8)))  __bf16 v8bf;
typedef __attribute__((ext_vector_type(8)))  float  v8f;

typedef int v4i_ __attribute__((vector_size(16)));
typedef __attribute__((address_space(1))) v4i_* gp128_t;   // global int4*
typedef __attribute__((address_space(3))) v4i_* lp128_t;   // LDS int4*

#if defined(__has_builtin)
#if __has_builtin(__builtin_amdgcn_global_load_async_to_lds_b128)
#define ASYNC_LDS 1
#endif
#if __has_builtin(__builtin_amdgcn_s_wait_asynccnt)
#define WAIT_ASYNC() __builtin_amdgcn_s_wait_asynccnt(0)
#endif
#endif
#ifndef ASYNC_LDS
#define ASYNC_LDS 0
#endif
#ifndef WAIT_ASYNC
#if ASYNC_LDS
#define WAIT_ASYNC() asm volatile("s_wait_asynccnt 0" ::: "memory")
#else
#define WAIT_ASYNC() ((void)0)
#endif
#endif

__device__ __forceinline__ unsigned short f32_to_bf16(float f) {
  unsigned int u = __builtin_bit_cast(unsigned int, f);
  u += 0x7FFFu + ((u >> 16) & 1u);          // round-to-nearest-even
  return (unsigned short)(u >> 16);
}

__device__ __forceinline__ v16bf cat8(v8bf lo, v8bf hi) {
  return __builtin_shufflevector(lo, hi, 0,1,2,3,4,5,6,7,8,9,10,11,12,13,14,15);
}

__device__ __forceinline__ v8f wmma_bf16(v16bf a, v16bf b, v8f c) {
  // 8 args: (neg_a, A, neg_b, B, c_mod, C, reuse_a, reuse_b)
  return __builtin_amdgcn_wmma_f32_16x16x32_bf16(false, a, false, b, (short)0, c,
                                                 false, false);
}

// A-fragment (16x32 bf16): lane<16 holds row=lane, K {0..15};
// lane>=16 holds row=lane-16, K {16..31}. Two 16B loads per lane.
__device__ __forceinline__ v16bf load_a_frag(const unsigned short* rowPtr, int laneH) {
  v8bf lo = *(const v8bf*)(rowPtr + laneH * 16);
  v8bf hi = *(const v8bf*)(rowPtr + 8 + laneH * 16);
  return cat8(lo, hi);
}

// Stage 16 bytes of B-slab global -> LDS (async-to-LDS on gfx1250).
__device__ __forceinline__ void stage16(const unsigned short* g, unsigned short* l) {
#if ASYNC_LDS
  __builtin_amdgcn_global_load_async_to_lds_b128((gp128_t)g, (lp128_t)l, 0, 0);
#else
  *(v8bf*)l = *(const v8bf*)g;
#endif
}

// ---------------------------------------------------------------------------
// fp32 -> bf16 bulk convert (4 elems / thread)
// ---------------------------------------------------------------------------
__global__ __launch_bounds__(256)
void k_f32_to_bf16(const float* __restrict__ src, unsigned short* __restrict__ dst,
                   int n4) {
  int i = blockIdx.x * 256 + threadIdx.x;
  if (i >= n4) return;
  const float4 f = ((const float4*)src)[i];
  ushort4 o;
  o.x = f32_to_bf16(f.x); o.y = f32_to_bf16(f.y);
  o.z = f32_to_bf16(f.z); o.w = f32_to_bf16(f.w);
  ((ushort4*)dst)[i] = o;
}

// ---------------------------------------------------------------------------
// C[M,N] = alpha * (A[M,K] @ Bsrc[N,K]^T + bias[N])
// Block = 8 waves; block tile 128(M) x 64(N); wave tile 16x64 (4 WMMA accs
// sharing one A fragment per k-step). B k-slab (64x32 bf16, 4 KB) double-
// buffered in LDS via async-to-LDS; last k-step peeled so the steady-state
// loop stages unconditionally (branch-free backedge).
// ---------------------------------------------------------------------------
__global__ __launch_bounds__(256)
void k_gemm_nt(const unsigned short* __restrict__ A,
               const unsigned short* __restrict__ Bsrc,
               const float* __restrict__ bias,
               float* __restrict__ C,
               int M, int N, int K, float alpha) {
  __shared__ unsigned short sB[2][64 * 32];
  const int tid   = threadIdx.x;
  const int wave  = tid >> 5;
  const int lane  = tid & 31;
  const int laneR = lane & 15;
  const int laneH = lane >> 4;
  const int nBase = blockIdx.x * 64;
  const int m0    = blockIdx.y * 128 + wave * 16;

  // staging coords: thread t -> slab row t/4 (0..63), 16B chunk (t%4)
  const int srow = tid >> 2;
  const int scol = (tid & 3) * 8;

  v8f acc[4] = {};
  const unsigned short* aRow  = A + (size_t)(m0 + laneR) * K;
  const unsigned short* bRowG = Bsrc + (size_t)(nBase + srow) * K + scol;

  stage16(bRowG, &sB[0][srow * 32 + scol]);
  WAIT_ASYNC();
  __syncthreads();

  int cur = 0;
  int k0  = 0;
  for (; k0 < K - 32; k0 += 32) {
    stage16(bRowG + k0 + 32, &sB[cur ^ 1][srow * 32 + scol]);

    __builtin_prefetch(aRow + k0 + 256, 0, 1);     // global_prefetch_b8

    v16bf a = load_a_frag(aRow + k0, laneH);
    const unsigned short* bb = &sB[cur][0];
#pragma unroll
    for (int j = 0; j < 4; ++j) {
      v16bf b = *(const v16bf*)(bb + (j * 16 + laneR) * 32 + laneH * 16);
      acc[j] = wmma_bf16(a, b, acc[j]);
    }

    WAIT_ASYNC();
    __syncthreads();
    cur ^= 1;
  }
  { // final k-step: no staging, no trailing barrier needed
    v16bf a = load_a_frag(aRow + k0, laneH);
    const unsigned short* bb = &sB[cur][0];
#pragma unroll
    for (int j = 0; j < 4; ++j) {
      v16bf b = *(const v16bf*)(bb + (j * 16 + laneR) * 32 + laneH * 16);
      acc[j] = wmma_bf16(a, b, acc[j]);
    }
  }

#pragma unroll
  for (int j = 0; j < 4; ++j) {
    const float bv = bias ? bias[nBase + j * 16 + laneR] : 0.0f;
#pragma unroll
    for (int i = 0; i < 8; ++i) {
      int m = m0 + i + laneH * 8;
      C[(size_t)m * N + nBase + j * 16 + laneR] = alpha * (acc[j][i] + bv);
    }
  }
}

// ---------------------------------------------------------------------------
// XPOS rotary + split heads + bf16: (B,T,E) f32 -> (B*H, T, HD) bf16
// ---------------------------------------------------------------------------
__global__ __launch_bounds__(256)
void k_xpos_split(const float* __restrict__ src, unsigned short* __restrict__ dst,
                  int downscale) {
  int p = blockIdx.x * 256 + threadIdx.x;            // pair id, B*T*E/2 total
  int e2 = p & (E_ / 2 - 1);
  int t  = (p >> 9) & (T_ - 1);
  int b  = p >> 19;
  int e  = e2 * 2;
  int h  = e >> 6;
  int d  = e & (HD_ - 1);
  int j  = d >> 1;                                   // 0..31

  size_t si = ((size_t)(b * T_ + t)) * E_ + e;
  float x1 = src[si];
  float x2 = src[si + 1];

  float inv_freq = __powf(10000.0f, -(float)j / 32.0f);
  float ang  = (float)t * inv_freq;
  float base = (2.0f * (float)j + 0.4f * (float)HD_) / (1.4f * (float)HD_);
  float ex   = (float)(t - T_ / 2) / 512.0f;         // pos / SCALE_BASE
  if (downscale) ex = -ex;
  float scl = __powf(base, ex);
  float s = __sinf(ang) * scl;
  float c = __cosf(ang) * scl;

  float o1 = x1 * c - x2 * s;
  float o2 = x2 * c + x1 * s;

  size_t o = ((size_t)(b * H_ + h) * T_ + t) * HD_ + d;
  dst[o]     = f32_to_bf16(o1);
  dst[o + 1] = f32_to_bf16(o2);
}

// ---------------------------------------------------------------------------
// V: (B,T,E) f32 -> transposed (B*H, HD, T) bf16 (PV B-frags become contiguous)
// ---------------------------------------------------------------------------
__global__ __launch_bounds__(256)
void k_v_transpose(const float* __restrict__ src, unsigned short* __restrict__ dst) {
  int i = blockIdx.x * 256 + threadIdx.x;            // B*T*E total
  int e = i & (E_ - 1);
  int t = (i >> 10) & (T_ - 1);
  int b = i >> 20;
  int h = e >> 6, d = e & (HD_ - 1);
  dst[((size_t)(b * H_ + h) * HD_ + d) * T_ + t] = f32_to_bf16(src[i]);
}

// ---------------------------------------------------------------------------
// Scores + mask + rel_pos + softmax, one (bh, 16-query) tile per block.
// 8 waves each cover 8 key tiles (QK^T via 2 WMMAs / tile, K=HD=64).
// Scores in 64 KB LDS; row reductions via __shfl_xor within a half-wave.
// Streaming traffic (rel_pos in, fp32 probs out) is non-temporal.
// ---------------------------------------------------------------------------
__global__ __launch_bounds__(256)
void k_attn_softmax(const unsigned short* __restrict__ qbh,
                    const unsigned short* __restrict__ kbh,
                    const float* __restrict__ mask,
                    const float* __restrict__ rel,
                    float* __restrict__ probs_out,
                    unsigned short* __restrict__ probs_bf) {
  __shared__ float sS[16 * T_];                      // 64 KB

  const int tid   = threadIdx.x;
  const int wave  = tid >> 5;
  const int lane  = tid & 31;
  const int laneR = lane & 15;
  const int laneH = lane >> 4;

  const int bid = blockIdx.x;
  const int qt  = bid & (T_ / 16 - 1);
  const int bh  = bid >> 6;
  const int b   = bh >> 4;
  const int h   = bh & (H_ - 1);
  const int q0  = qt * 16;

  const unsigned short* qbase = qbh + ((size_t)bh * T_ + q0) * HD_;
  const unsigned short* kbase = kbh + (size_t)bh * T_ * HD_;

  // Q fragments for K-dim 0..31 and 32..63 (reused across all key tiles)
  const unsigned short* aRow = qbase + laneR * HD_;
  v16bf a0 = load_a_frag(aRow,      laneH);
  v16bf a1 = load_a_frag(aRow + 32, laneH);

  for (int kt = wave; kt < T_ / 16; kt += 8) {
    const unsigned short* kRow = kbase + (size_t)(kt * 16 + laneR) * HD_;
    // B = K^T: per-lane column of B == contiguous row of K
    v16bf b0 = *(const v16bf*)(kRow + laneH * 16);
    v16bf b1 = *(const v16bf*)(kRow + 32 + laneH * 16);
    v8f acc = {};
    acc = wmma_bf16(a0, b0, acc);
    acc = wmma_bf16(a1, b1, acc);
#pragma unroll
    for (int i = 0; i < 8; ++i) {
      int m    = i + laneH * 8;
      int qrow = q0 + m;
      int kcol = kt * 16 + laneR;
      float rp = __builtin_nontemporal_load(&rel[((size_t)bh * T_ + qrow) * T_ + kcol]);
      float sc = acc[i] + mask[(size_t)qrow * T_ + kcol] + rp;
      sS[m * T_ + kcol] = sc;
    }
  }
  __syncthreads();

  // softmax: row r handled by 16 consecutive threads (one half-wave)
  const int r  = tid >> 4;
  const int c0 = tid & 15;

  float mx = -1e30f;
  for (int c = c0; c < T_; c += 16) mx = fmaxf(mx, sS[r * T_ + c]);
#pragma unroll
  for (int off = 8; off; off >>= 1) mx = fmaxf(mx, __shfl_xor(mx, off, 16));

  float sum = 0.0f;
  for (int c = c0; c < T_; c += 16) {
    float ev = __expf(sS[r * T_ + c] - mx);
    sS[r * T_ + c] = ev;
    sum += ev;
  }
#pragma unroll
  for (int off = 8; off; off >>= 1) sum += __shfl_xor(sum, off, 16);
  const float inv = 1.0f / sum;

  const size_t obase = ((size_t)(h * B_ + b) * T_ + (q0 + r)) * T_;  // (H,B,T,T)
  const size_t pbase = ((size_t)bh * T_ + (q0 + r)) * T_;            // (BH,T,T)
  for (int c = c0; c < T_; c += 16) {
    float pv = sS[r * T_ + c] * inv;
    __builtin_nontemporal_store(pv, &probs_out[obase + c]);
    probs_bf[pbase + c] = f32_to_bf16(pv);
  }
}

// ---------------------------------------------------------------------------
// ctx = P @ V : per (bh, 16-query) block; 4 waves = 4 HD tiles of 16.
// Causality: prob columns > q0+15 are exactly 0 => K loop ends at q0+16.
// ---------------------------------------------------------------------------
__global__ __launch_bounds__(128)
void k_pv(const unsigned short* __restrict__ probs_bf,
          const unsigned short* __restrict__ vt,
          unsigned short* __restrict__ ctx) {
  const int tid   = threadIdx.x;
  const int wave  = tid >> 5;
  const int lane  = tid & 31;
  const int laneR = lane & 15;
  const int laneH = lane >> 4;

  const int bid = blockIdx.x;
  const int qt  = bid & (T_ / 16 - 1);
  const int bh  = bid >> 6;
  const int b   = bh >> 4;
  const int h   = bh & (H_ - 1);
  const int q0  = qt * 16;
  const int d0  = wave * 16;

  const unsigned short* pbase = probs_bf + ((size_t)bh * T_ + q0) * T_;
  const unsigned short* vbase = vt + ((size_t)bh * HD_ + d0) * T_;

  v8f acc = {};
  const int kend = (q0 + 16 + 31) & ~31;             // causal upper bound
  for (int k0 = 0; k0 < kend; k0 += 32) {
    const unsigned short* aRow = pbase + (size_t)laneR * T_ + k0;
    v16bf a = load_a_frag(aRow, laneH);
    // B[kk, d] = V[kk, d] = vt[d, kk]: per-lane contiguous run of vt row
    v16bf bfrag = *(const v16bf*)(vbase + (size_t)laneR * T_ + k0 + laneH * 16);
    acc = wmma_bf16(a, bfrag, acc);
  }
#pragma unroll
  for (int i = 0; i < 8; ++i) {
    int q = q0 + i + laneH * 8;
    ctx[((size_t)(b * T_ + q)) * E_ + h * HD_ + d0 + laneR] = f32_to_bf16(acc[i]);
  }
}

// ---------------------------------------------------------------------------
// Host-side orchestration
// ---------------------------------------------------------------------------
extern "C" void kernel_launch(void* const* d_in, const int* in_sizes, int n_in,
                              void* d_out, int out_size, void* d_ws, size_t ws_size,
                              hipStream_t stream) {
  const float* q_in  = (const float*)d_in[0];
  const float* k_in  = (const float*)d_in[1];
  const float* v_in  = (const float*)d_in[2];
  const float* rel   = (const float*)d_in[3];
  const float* Wq    = (const float*)d_in[4];
  const float* bq    = (const float*)d_in[5];
  const float* Wk    = (const float*)d_in[6];
  const float* bk    = (const float*)d_in[7];
  const float* Wv    = (const float*)d_in[8];
  const float* bv    = (const float*)d_in[9];
  const float* Wo    = (const float*)d_in[10];
  const float* bo    = (const float*)d_in[11];
  const float* mask  = (const float*)d_in[12];

  float* out_proj  = (float*)d_out;                        // (B,T,E)
  float* out_probs = (float*)d_out + (size_t)B_ * T_ * E_; // (H,B,T,T)

  const size_t MB = (size_t)1 << 20;
  uint8_t* ws = (uint8_t*)d_ws;
  unsigned short* xq_bf  = (unsigned short*)(ws + 0 * MB);    // 4 MB each
  unsigned short* xk_bf  = (unsigned short*)(ws + 4 * MB);
  unsigned short* xv_bf  = (unsigned short*)(ws + 8 * MB);
  unsigned short* wq_bf  = (unsigned short*)(ws + 12 * MB);   // 2 MB each
  unsigned short* wk_bf  = (unsigned short*)(ws + 14 * MB);
  unsigned short* wv_bf  = (unsigned short*)(ws + 16 * MB);
  unsigned short* wo_bf  = (unsigned short*)(ws + 18 * MB);
  float*          qproj  = (float*)(ws + 20 * MB);            // 8 MB each
  float*          kproj  = (float*)(ws + 28 * MB);
  float*          vproj  = (float*)(ws + 36 * MB);
  unsigned short* qbh    = (unsigned short*)(ws + 44 * MB);   // 4 MB each
  unsigned short* kbh    = (unsigned short*)(ws + 48 * MB);
  unsigned short* vt     = (unsigned short*)(ws + 52 * MB);
  unsigned short* pbf    = (unsigned short*)(ws + 56 * MB);   // 64 MB
  unsigned short* ctx    = (unsigned short*)(ws + 120 * MB);  // 4 MB

  const int nAct = B_ * T_ * E_;   // 2M
  const int nW   = E_ * E_;        // 1M

  // fp32 -> bf16 conversions
  k_f32_to_bf16<<<nAct / 4 / 256, 256, 0, stream>>>(q_in, xq_bf, nAct / 4);
  k_f32_to_bf16<<<nAct / 4 / 256, 256, 0, stream>>>(k_in, xk_bf, nAct / 4);
  k_f32_to_bf16<<<nAct / 4 / 256, 256, 0, stream>>>(v_in, xv_bf, nAct / 4);
  k_f32_to_bf16<<<nW / 4 / 256, 256, 0, stream>>>(Wq, wq_bf, nW / 4);
  k_f32_to_bf16<<<nW / 4 / 256, 256, 0, stream>>>(Wk, wk_bf, nW / 4);
  k_f32_to_bf16<<<nW / 4 / 256, 256, 0, stream>>>(Wv, wv_bf, nW / 4);
  k_f32_to_bf16<<<nW / 4 / 256, 256, 0, stream>>>(Wo, wo_bf, nW / 4);

  // projections: (B*T, E) @ W^T + b ; q pre-scaled by HD^-0.5
  const int M = B_ * T_;
  dim3 gGemm(E_ / 64, M / 128);
  k_gemm_nt<<<gGemm, 256, 0, stream>>>(xq_bf, wq_bf, bq, qproj, M, E_, E_, 0.125f);
  k_gemm_nt<<<gGemm, 256, 0, stream>>>(xk_bf, wk_bf, bk, kproj, M, E_, E_, 1.0f);
  k_gemm_nt<<<gGemm, 256, 0, stream>>>(xv_bf, wv_bf, bv, vproj, M, E_, E_, 1.0f);

  // XPOS rotary + head split (q: upscale, k: downscale); v transpose
  k_xpos_split<<<nAct / 2 / 256, 256, 0, stream>>>(qproj, qbh, 0);
  k_xpos_split<<<nAct / 2 / 256, 256, 0, stream>>>(kproj, kbh, 1);
  k_v_transpose<<<nAct / 256, 256, 0, stream>>>(vproj, vt);

  // scores + softmax (writes both prob outputs)
  k_attn_softmax<<<B_ * H_ * (T_ / 16), 256, 0, stream>>>(qbh, kbh, mask, rel,
                                                          out_probs, pbf);

  // context = P @ V
  k_pv<<<B_ * H_ * (T_ / 16), 128, 0, stream>>>(pbf, vt, ctx);

  // output projection
  k_gemm_nt<<<gGemm, 256, 0, stream>>>(ctx, wo_bf, bo, out_proj, M, E_, E_, 1.0f);
}
